// PredictiveCodingCorrector_1236950581815
// MI455X (gfx1250) — matrix operand order repaired and checked by reading, not compiled, not run on hardware
//
#include <hip/hip_runtime.h>
#include <hip/hip_bf16.h>
#include <math.h>

typedef _Float16 v16h __attribute__((ext_vector_type(16)));
typedef _Float16 h8   __attribute__((ext_vector_type(8)));
typedef float    v8f  __attribute__((ext_vector_type(8)));

#define BB      8
#define TT      4096
#define NS      512
#define DD      1024
#define HID     128
#define NROWS   (BB * TT)      // 32768
#define LDA     1032           // padded f16 stride for 1024-wide LDS tiles
#define LDH     136            // padded f16 stride for 128-wide LDS tiles
#define CHUNK   64
#define NCHUNK  (TT / CHUNK)   // 64
#define V8F_ZERO (v8f){0.f,0.f,0.f,0.f,0.f,0.f,0.f,0.f}

__device__ __forceinline__ v8f wmma_f16(v16h a, v16h b, v8f c) {
  return __builtin_amdgcn_wmma_f32_16x16x32_f16(false, a, false, b, (short)0, c,
                                                false, false);
}

// A fragment (16x32 f16) from row-major LDS tile.
// ISA layout: lane holds row = lane&15; VGPR v0..3 -> K = (lane>>4)*8 + 0..7,
// v4..7 -> K = 16 + (lane>>4)*8 + 0..7  => two contiguous 16B LDS reads.
__device__ __forceinline__ v16h load_a_tile(const _Float16* base, int ld,
                                            int lane, int kt) {
  const _Float16* p = base + (lane & 15) * ld + ((lane >> 4) << 3) + (kt << 5);
  union { v16h v; h8 h[2]; } u;
  u.h[0] = *(const h8*)(p);
  u.h[1] = *(const h8*)(p + 16);
  return u.v;
}

// B fragment (32x16 f16) from pre-packed global weights: 32 contiguous bytes per lane.
__device__ __forceinline__ v16h load_b_tile(const _Float16* __restrict__ packed,
                                            int tile, int lane) {
  return *(const v16h*)(packed + ((size_t)tile << 9) + (lane << 4));
}

__device__ __forceinline__ float gelu_exact(float x) {
  return 0.5f * x * (1.0f + erff(x * 0.70710678118654752f));
}

__device__ __forceinline__ float sigmoidf_(float x) {
  return 1.0f / (1.0f + expf(-x));
}

// Pack f32 row-major (K x N) weights into f16 WMMA-B tile order.
// Tile (nt,kt) holds B[kt*32 .. +31][nt*16 .. +15]; within a tile, element
// order is lane-major matching the B operand register layout:
//   lane = n_loc + 16*((kk>>3)&1),  j = 8*(kk>>4) + (kk&7)   (kk = k within tile)
__global__ void pack_b_weights(const float* __restrict__ src,
                               _Float16* __restrict__ dst, int K, int N) {
  int KT = K >> 5;
  int total = K * N;
  for (int p = blockIdx.x * blockDim.x + threadIdx.x; p < total;
       p += gridDim.x * blockDim.x) {
    int tile = p >> 9;
    int q = p & 511;
    int lane = q >> 4;
    int j = q & 15;
    int nt = tile / KT;
    int kt = tile - nt * KT;
    int kk = ((j >> 3) << 4) | ((lane >> 4) << 3) | (j & 7);
    int k = (kt << 5) + kk;
    int n = (nt << 4) + (lane & 15);
    dst[p] = (_Float16)src[k * N + n];
  }
}

__global__ void gate_kernel(const float* __restrict__ log_gain,
                            float* __restrict__ gateOut) {
  int i = blockIdx.x * blockDim.x + threadIdx.x;
  if (i < NS) gateOut[i] = sigmoidf_(log_gain[i]);
}

// Fused: trig features -> GEMM1(+GELU) -> GEMM2 -> err -> GEMM3(+GELU) -> GEMM4
// One block = 16 rows, 8 waves; activations live in LDS as f16.
__global__ __launch_bounds__(256) void fused_mlp_kernel(
    const float* __restrict__ theta, const float* __restrict__ content,
    const unsigned char* __restrict__ mask,
    const float* __restrict__ fb1, const float* __restrict__ fb2,
    const float* __restrict__ eb1, const float* __restrict__ eb2,
    const _Float16* __restrict__ W1f, const _Float16* __restrict__ W2f,
    const _Float16* __restrict__ W1e, const _Float16* __restrict__ W2e,
    float* __restrict__ errOut, float* __restrict__ deltaOut) {
  __shared__ __align__(16) _Float16 sA[16 * LDA];  // pos_feat, then masked err
  __shared__ __align__(16) _Float16 sH[16 * LDH];  // hidden activations
  __shared__ float smask[16];

  const int tid  = threadIdx.x;
  const int lane = tid & 31;
  const int wave = tid >> 5;
  const int r0   = blockIdx.x << 4;

  if (tid < 16) smask[tid] = mask[r0 + tid] ? 1.0f : 0.0f;

  // Stage 1: pos_feat[row][col]; col -> (head = col/64, cos if (col%64)<32)
  for (int e = tid; e < 16 * 1024; e += 256) {
    int row = e >> 10;
    int col = e & 1023;
    float th = theta[(size_t)(r0 + row) * NS + ((col >> 6) << 5) + (col & 31)];
    float v = ((col & 63) < 32) ? cosf(th) : sinf(th);
    sA[row * LDA + col] = (_Float16)v;
  }
  __syncthreads();

  const int nloc  = lane & 15;
  const int rbase = (lane >> 4) << 3;

  // GEMM1: (16x1024) @ fw1 -> 16x128, +bias, GELU -> sH. Wave w owns N-tile w.
  {
    v8f acc = V8F_ZERO;
    for (int kt = 0; kt < 32; ++kt)
      acc = wmma_f16(load_a_tile(sA, LDA, lane, kt),
                     load_b_tile(W1f, wave * 32 + kt, lane), acc);
    int n = (wave << 4) + nloc;
    float b = fb1[n];
#pragma unroll
    for (int r = 0; r < 8; ++r)
      sH[(rbase + r) * LDH + n] = (_Float16)gelu_exact(acc[r] + b);
  }
  __syncthreads();

  // GEMM2: (16x128) @ fw2 -> 16x1024; err_raw out; masked err -> sA (f16)
  for (int i = 0; i < 8; ++i) {
    int nt = (wave << 3) + i;
    v8f acc = V8F_ZERO;
    for (int kt = 0; kt < 4; ++kt)
      acc = wmma_f16(load_a_tile(sH, LDH, lane, kt),
                     load_b_tile(W2f, (nt << 2) + kt, lane), acc);
    int n = (nt << 4) + nloc;
    float b = fb2[n];
#pragma unroll
    for (int r = 0; r < 8; ++r) {
      int row = rbase + r;
      size_t gi = (size_t)(r0 + row) * DD + n;
      float er = content[gi] - (acc[r] + b);
      errOut[gi] = er;
      sA[row * LDA + n] = (_Float16)(er * smask[row]);
    }
  }
  __syncthreads();

  // GEMM3: err(16x1024) @ ew1 -> 16x128, +bias, GELU -> sH
  {
    v8f acc = V8F_ZERO;
    for (int kt = 0; kt < 32; ++kt)
      acc = wmma_f16(load_a_tile(sA, LDA, lane, kt),
                     load_b_tile(W1e, wave * 32 + kt, lane), acc);
    int n = (wave << 4) + nloc;
    float b = eb1[n];
#pragma unroll
    for (int r = 0; r < 8; ++r)
      sH[(rbase + r) * LDH + n] = (_Float16)gelu_exact(acc[r] + b);
  }
  __syncthreads();

  // GEMM4: (16x128) @ ew2 -> 16x512 = delta_theta -> deltaOut (theta_hat slot)
  for (int i = 0; i < 4; ++i) {
    int nt = (wave << 2) + i;
    v8f acc = V8F_ZERO;
    for (int kt = 0; kt < 4; ++kt)
      acc = wmma_f16(load_a_tile(sH, LDH, lane, kt),
                     load_b_tile(W2e, (nt << 2) + kt, lane), acc);
    int n = (nt << 4) + nloc;
    float b = eb2[n];
#pragma unroll
    for (int r = 0; r < 8; ++r)
      deltaOut[(size_t)(r0 + rbase + r) * NS + n] = acc[r] + b;
  }
}

// Scan pass 1: local scan within each 64-step chunk (d_{-1}=0), in place.
__global__ void scan_pass1(const float* __restrict__ log_gain,
                           float* __restrict__ dbuf,
                           float* __restrict__ carries) {
  int idx = blockIdx.x * blockDim.x + threadIdx.x;  // B*NS*NCHUNK threads
  int c = idx & 511;
  int b = (idx >> 9) & 7;
  int chunk = idx >> 12;
  float g = sigmoidf_(log_gain[c]);
  float d = 0.0f;
  size_t base = ((size_t)b * TT + chunk * CHUNK) * NS + c;
  for (int i = 0; i < CHUNK; ++i) {
    size_t off = base + (size_t)i * NS;
    d = g * d + dbuf[off];
    dbuf[off] = d;
  }
  carries[(size_t)chunk * (BB * NS) + b * NS + c] = d;
}

// Scan pass 2: exclusive scan over chunk carries with factor g^CHUNK, in place.
__global__ void scan_pass2(const float* __restrict__ log_gain,
                           float* __restrict__ carries) {
  int idx = blockIdx.x * blockDim.x + threadIdx.x;  // B*NS threads
  int c = idx & 511;
  float g = sigmoidf_(log_gain[c]);
  float gL = g;
  for (int s = 0; s < 6; ++s) gL *= gL;  // g^64
  float acc = 0.0f;
  for (int ch = 0; ch < NCHUNK; ++ch) {
    size_t o = (size_t)ch * (BB * NS) + idx;
    float v = carries[o];
    carries[o] = acc;  // d at end of previous chunk
    acc = gL * acc + v;
  }
}

// Scan pass 3: apply carry (d_t += g^{i+1} * carry_in) and add theta residual.
__global__ void scan_pass3(const float* __restrict__ log_gain,
                           const float* __restrict__ theta,
                           float* __restrict__ dbuf,
                           const float* __restrict__ carries) {
  int idx = blockIdx.x * blockDim.x + threadIdx.x;
  int c = idx & 511;
  int b = (idx >> 9) & 7;
  int chunk = idx >> 12;
  float g = sigmoidf_(log_gain[c]);
  float carry = carries[(size_t)chunk * (BB * NS) + b * NS + c];
  float gp = g;
  size_t base = ((size_t)b * TT + chunk * CHUNK) * NS + c;
  for (int i = 0; i < CHUNK; ++i) {
    size_t off = base + (size_t)i * NS;
    float d = dbuf[off] + gp * carry;
    gp *= g;
    dbuf[off] = theta[off] + d;
  }
}

extern "C" void kernel_launch(void* const* d_in, const int* in_sizes, int n_in,
                              void* d_out, int out_size, void* d_ws,
                              size_t ws_size, hipStream_t stream) {
  (void)in_sizes; (void)n_in; (void)out_size; (void)ws_size;
  const float* theta    = (const float*)d_in[0];
  const float* content  = (const float*)d_in[1];
  const unsigned char* mask = (const unsigned char*)d_in[2];  // jnp bool_
  const float* fw1 = (const float*)d_in[3];
  const float* fb1 = (const float*)d_in[4];
  const float* fw2 = (const float*)d_in[5];
  const float* fb2 = (const float*)d_in[6];
  const float* ew1 = (const float*)d_in[7];
  const float* eb1 = (const float*)d_in[8];
  const float* ew2 = (const float*)d_in[9];
  const float* eb2 = (const float*)d_in[10];
  const float* log_gain = (const float*)d_in[11];

  float* out      = (float*)d_out;
  float* thetaHat = out;                              // also delta scratch
  float* errOut   = out + (size_t)NROWS * NS;
  float* gateOut  = errOut + (size_t)NROWS * DD;

  // workspace: carries (64*4096 f32 = 1 MiB), then packed f16 weights (~0.9 MB)
  float* carries = (float*)d_ws;
  _Float16* wbase = (_Float16*)((char*)d_ws + (size_t)NCHUNK * BB * NS * 4);
  _Float16* W1f = wbase;                 // 8*32  tiles = 131072 halves
  _Float16* W2f = W1f + 131072;          // 64*4  tiles
  _Float16* W1e = W2f + 131072;          // 8*32  tiles
  _Float16* W2e = W1e + 131072;          // 32*4  tiles = 65536 halves

  pack_b_weights<<<128, 256, 0, stream>>>(fw1, W1f, 1024, 128);
  pack_b_weights<<<128, 256, 0, stream>>>(fw2, W2f, 128, 1024);
  pack_b_weights<<<128, 256, 0, stream>>>(ew1, W1e, 1024, 128);
  pack_b_weights<<<64, 256, 0, stream>>>(ew2, W2e, 128, 512);
  gate_kernel<<<2, 256, 0, stream>>>(log_gain, gateOut);

  fused_mlp_kernel<<<NROWS / 16, 256, 0, stream>>>(
      theta, content, mask, fb1, fb2, eb1, eb2, W1f, W2f, W1e, W2e, errOut,
      thetaHat);

  scan_pass1<<<(BB * NS * NCHUNK) / 256, 256, 0, stream>>>(log_gain, thetaHat,
                                                           carries);
  scan_pass2<<<(BB * NS) / 256, 256, 0, stream>>>(log_gain, carries);
  scan_pass3<<<(BB * NS * NCHUNK) / 256, 256, 0, stream>>>(log_gain, theta,
                                                           thetaHat, carries);
}
